// MultiHeadAttentionBlock_79517024518931
// MI455X (gfx1250) — matrix operand order repaired and checked by reading, not compiled, not run on hardware
//
#include <hip/hip_runtime.h>
#include <hip/hip_bf16.h>
#include <math.h>

// ---- sizes ----
#define Dm   1024
#define Hh   16
#define Ss   2048
#define Bb   2
#define HD   64            // head dim
#define Mrows (Bb*Ss)      // 4096

typedef __attribute__((ext_vector_type(16))) __bf16 v16bf;
typedef __attribute__((ext_vector_type(8)))  __bf16 v8bf;
typedef __attribute__((ext_vector_type(8)))  float  v8f;

// ---------------- fragment loaders ----------------
// A-operand (16x32 bf16): lane r = M row, half = lane>>4.
// Hardware striping: low-half lanes hold K = 0..7 and 16..23 (VGPR pairs),
// high-half lanes hold K = 8..15 and 24..31  -> two 16B chunks per lane.
__device__ __forceinline__ v16bf load_a_frag(const __bf16* rowp, int k0, int half) {
    const __bf16* p = rowp + k0 + half * 8;
    union { v8bf h[2]; v16bf v; } u;
    u.h[0] = *reinterpret_cast<const v8bf*>(p);
    u.h[1] = *reinterpret_cast<const v8bf*>(p + 16);
    return u.v;
}
// B-operand (32x16 bf16): lane r = N col, same K striping as A:
// low-half lanes hold K = 0..7 and 16..23, high-half lanes K = 8..15 and 24..31.
__device__ __forceinline__ v16bf load_b_frag(const __bf16* rowp, int k0, int half) {
    const __bf16* p = rowp + k0 + half * 8;
    union { v8bf h[2]; v16bf v; } u;
    u.h[0] = *reinterpret_cast<const v8bf*>(p);
    u.h[1] = *reinterpret_cast<const v8bf*>(p + 16);
    return u.v;
}

#define WMMA_BF16(a, b, c) \
    __builtin_amdgcn_wmma_f32_16x16x32_bf16(false, (a), false, (b), (short)0, (c), false, false)

// ---------------- fp32 -> bf16 convert ----------------
__global__ __launch_bounds__(256) void cvt_f32_bf16(const float* __restrict__ src,
                                                    __bf16* __restrict__ dst, int n) {
    int i = blockIdx.x * 256 + threadIdx.x;
    if (i < n) dst[i] = (__bf16)src[i];
}

// ---------------- QKV GEMM:  qkv[m,n] = sum_k x[m,k] * Ww[n,k] + Wb[n] ----------------
// Per-wave 32(M) x 64(N) tile: two A fragments share each B fragment.
// Scatter into q/k [B,H,S,64] and vT [B,H,64,S].
__global__ __launch_bounds__(256) void qkv_gemm(const __bf16* __restrict__ X,
                                                const __bf16* __restrict__ W,
                                                const float*  __restrict__ Wb,
                                                __bf16* __restrict__ qb,
                                                __bf16* __restrict__ kb,
                                                __bf16* __restrict__ vT) {
    const int wave = threadIdx.x >> 5;
    const int lane = threadIdx.x & 31;
    const int gw   = blockIdx.x * 8 + wave;         // 0 .. 6143
    const int NT   = (3 * Dm) / 64;                 // 48
    const int mt   = gw / NT;                       // 0..127 (tiles of 32 rows)
    const int nt   = gw % NT;
    const int m0 = mt * 32, n0 = nt * 64;
    const int r = lane & 15, half = lane >> 4;

    const __bf16* arow0 = X + (size_t)(m0 + r) * Dm;
    const __bf16* arow1 = X + (size_t)(m0 + 16 + r) * Dm;
    v8f acc0[4] = {};
    v8f acc1[4] = {};
    for (int k0 = 0; k0 < Dm; k0 += 32) {
        v16bf a0 = load_a_frag(arow0, k0, half);
        v16bf a1 = load_a_frag(arow1, k0, half);
#pragma unroll
        for (int nc = 0; nc < 4; ++nc) {
            v16bf bf = load_b_frag(W + (size_t)(n0 + nc * 16 + r) * Dm, k0, half);
            acc0[nc] = WMMA_BF16(a0, bf, acc0[nc]);
            acc1[nc] = WMMA_BF16(a1, bf, acc1[nc]);
        }
    }
    // C layout: vgpr g -> row (m-base) + g + 8*half ; col n0 + nc*16 + r
#pragma unroll
    for (int nc = 0; nc < 4; ++nc) {
        const int n = n0 + nc * 16 + r;
        const int seg = n >> 10;          // 0=q 1=k 2=v
        const int rr  = n & 1023;
        const int h   = rr >> 6;
        const int dim = rr & 63;
        const float bias = Wb[n];
#pragma unroll
        for (int mh = 0; mh < 2; ++mh) {
#pragma unroll
            for (int g = 0; g < 8; ++g) {
                const int m = m0 + mh * 16 + g + 8 * half;
                const int b = m >> 11;        // / S
                const int s = m & (Ss - 1);
                const size_t bh = (size_t)(b * Hh + h);
                const float av = (mh == 0) ? acc0[nc][g] : acc1[nc][g];
                const __bf16 val = (__bf16)(av + bias);
                if (seg == 0)      qb[(bh * Ss + s) * HD + dim] = val;
                else if (seg == 1) kb[(bh * Ss + s) * HD + dim] = val;
                else               vT[(bh * HD + dim) * Ss + s] = val;
            }
        }
    }
}

// ---------------- flash attention: one wave per (b,h, 16 query rows) ----------------
__global__ __launch_bounds__(32) void attn_kernel(const __bf16* __restrict__ qb,
                                                  const __bf16* __restrict__ kb,
                                                  const __bf16* __restrict__ vT,
                                                  const float*  __restrict__ rpb,
                                                  __bf16* __restrict__ ob) {
    __shared__ alignas(32) __bf16 Plds[16 * 32];    // P tile, row-major [16][32]
    const int lane = threadIdx.x;
    const int r = lane & 15, half = lane >> 4;
    const int qt    = blockIdx.x;                   // 0 .. 4095
    const int qtile = qt & (Ss / 16 - 1);           // 0..127
    const int bh    = qt >> 7;                      // b*H + h
    const int h     = bh & (Hh - 1);
    const int bidx  = bh >> 4;
    const int q0    = qtile * 16;

    const __bf16* Q = qb + (size_t)bh * Ss * HD;
    const __bf16* K = kb + (size_t)bh * Ss * HD;
    const __bf16* V = vT + (size_t)bh * HD * Ss;

    const __bf16* qrow = Q + (size_t)(q0 + r) * HD;
    const v16bf qfa0 = load_a_frag(qrow, 0, half);
    const v16bf qfa1 = load_a_frag(qrow, 32, half);

    v8f   O[4] = {};
    float mrow[8], lrow[8];
#pragma unroll
    for (int g = 0; g < 8; ++g) { mrow[g] = -__builtin_inff(); lrow[g] = 0.f; }

    const float scale = 0.03125f;                   // 1/sqrt(1024)
    const int nkb = ((q0 + 15) >> 5) + 1;           // causal key-block count (32 keys each)

    for (int kb32 = 0; kb32 < nkb; ++kb32) {
        const int j0 = kb32 * 32;
        // ---- scores: 16 x 32 (two 16x16 tiles) ----
        v8f s0 = {}, s1 = {};
        {
            const __bf16* krow0 = K + (size_t)(j0 + r) * HD;
            s0 = WMMA_BF16(qfa0, load_b_frag(krow0, 0, half), s0);
            s0 = WMMA_BF16(qfa1, load_b_frag(krow0, 32, half), s0);
            const __bf16* krow1 = K + (size_t)(j0 + 16 + r) * HD;
            s1 = WMMA_BF16(qfa0, load_b_frag(krow1, 0, half), s1);
            s1 = WMMA_BF16(qfa1, load_b_frag(krow1, 32, half), s1);
        }
        // ---- bias + causal mask + online softmax ----
        float mnew[8], alpha[8];
#pragma unroll
        for (int g = 0; g < 8; ++g) {
            const int i  = q0 + g + 8 * half;       // query row
            const int j  = j0 + r;                  // key col (tile0)
            const int j1 = j + 16;
            float v0 = s0[g] * scale + rpb[(size_t)(j  - i + (Ss - 1)) * Hh + h];
            float v1 = s1[g] * scale + rpb[(size_t)(j1 - i + (Ss - 1)) * Hh + h];
            if (j  > i) v0 = -__builtin_inff();
            if (j1 > i) v1 = -__builtin_inff();
            s0[g] = v0; s1[g] = v1;
            float mx = fmaxf(v0, v1);
#pragma unroll
            for (int off = 1; off < 16; off <<= 1) mx = fmaxf(mx, __shfl_xor(mx, off));
            mnew[g]  = fmaxf(mrow[g], mx);
            alpha[g] = __expf(mrow[g] - mnew[g]);
            mrow[g]  = mnew[g];
        }
#pragma unroll
        for (int g = 0; g < 8; ++g) {
            float p0 = __expf(s0[g] - mnew[g]);
            float p1 = __expf(s1[g] - mnew[g]);
            s0[g] = p0; s1[g] = p1;
            float ls = p0 + p1;
#pragma unroll
            for (int off = 1; off < 16; off <<= 1) ls += __shfl_xor(ls, off);
            lrow[g] = lrow[g] * alpha[g] + ls;
        }
#pragma unroll
        for (int nc = 0; nc < 4; ++nc)
#pragma unroll
            for (int g = 0; g < 8; ++g) O[nc][g] *= alpha[g];

        // ---- re-shuffle P (C layout) -> A-operand layout via LDS ----
        __syncthreads();
#pragma unroll
        for (int g = 0; g < 8; ++g) {
            const int row = g + 8 * half;
            Plds[row * 32 + r]      = (__bf16)s0[g];
            Plds[row * 32 + 16 + r] = (__bf16)s1[g];
        }
        __syncthreads();
        const v16bf pf = load_a_frag(&Plds[r * 32], 0, half);

        // ---- O += P @ V  (K = 32 keys, N = 64 dims) ----
#pragma unroll
        for (int nc = 0; nc < 4; ++nc) {
            v16bf vf = load_b_frag(V + (size_t)(nc * 16 + r) * Ss, j0, half);
            O[nc] = WMMA_BF16(pf, vf, O[nc]);
        }
    }

    // ---- normalize + store to o [B,S,D] in bf16 ----
#pragma unroll
    for (int g = 0; g < 8; ++g) {
        const float inv = 1.0f / lrow[g];
        const int s = q0 + g + 8 * half;
#pragma unroll
        for (int nc = 0; nc < 4; ++nc) {
            const int dim = nc * 16 + r;
            ob[((size_t)(bidx * Ss + s)) * Dm + h * HD + dim] = (__bf16)(O[nc][g] * inv);
        }
    }
}

// ---------------- projection + residual: Y = X + O @ Pw^T + pb ----------------
// Per-wave 32(M) x 64(N) tile, same structure as qkv_gemm.
__global__ __launch_bounds__(256) void proj_gemm(const __bf16* __restrict__ O,
                                                 const __bf16* __restrict__ Pw,
                                                 const float*  __restrict__ pb,
                                                 const float*  __restrict__ X,
                                                 float* __restrict__ Y) {
    const int wave = threadIdx.x >> 5;
    const int lane = threadIdx.x & 31;
    const int gw = blockIdx.x * 8 + wave;           // 0..2047
    const int mt = gw >> 4;                         // 0..127 (tiles of 32 rows)
    const int nt = gw & 15;
    const int m0 = mt * 32, n0 = nt * 64;
    const int r = lane & 15, half = lane >> 4;

    const __bf16* arow0 = O + (size_t)(m0 + r) * Dm;
    const __bf16* arow1 = O + (size_t)(m0 + 16 + r) * Dm;
    v8f acc0[4] = {};
    v8f acc1[4] = {};
    for (int k0 = 0; k0 < Dm; k0 += 32) {
        v16bf a0 = load_a_frag(arow0, k0, half);
        v16bf a1 = load_a_frag(arow1, k0, half);
#pragma unroll
        for (int nc = 0; nc < 4; ++nc) {
            v16bf bf = load_b_frag(Pw + (size_t)(n0 + nc * 16 + r) * Dm, k0, half);
            acc0[nc] = WMMA_BF16(a0, bf, acc0[nc]);
            acc1[nc] = WMMA_BF16(a1, bf, acc1[nc]);
        }
    }
#pragma unroll
    for (int nc = 0; nc < 4; ++nc) {
        const int n = n0 + nc * 16 + r;
        const float bias = pb[n];
#pragma unroll
        for (int mh = 0; mh < 2; ++mh) {
#pragma unroll
            for (int g = 0; g < 8; ++g) {
                const int m = m0 + mh * 16 + g + 8 * half;
                const size_t idx = (size_t)m * Dm + n;
                const float av = (mh == 0) ? acc0[nc][g] : acc1[nc][g];
                Y[idx] = X[idx] + av + bias;
            }
        }
    }
}

// ---------------- LayerNorm over D, in-place on Y ----------------
__global__ __launch_bounds__(256) void ln_kernel(float* __restrict__ Y,
                                                 const float* __restrict__ gam,
                                                 const float* __restrict__ bet) {
    const int row = blockIdx.x;
    float* y = Y + (size_t)row * Dm;
    float s = 0.f, s2 = 0.f;
    for (int i = threadIdx.x; i < Dm; i += 256) { float v = y[i]; s += v; s2 += v * v; }
#pragma unroll
    for (int off = 1; off < 32; off <<= 1) { s += __shfl_xor(s, off); s2 += __shfl_xor(s2, off); }
    __shared__ float rs[8], rs2[8];
    const int wave = threadIdx.x >> 5, lane = threadIdx.x & 31;
    if (lane == 0) { rs[wave] = s; rs2[wave] = s2; }
    __syncthreads();
    if (threadIdx.x == 0) {
        float a = 0.f, a2 = 0.f;
        for (int w = 0; w < 8; ++w) { a += rs[w]; a2 += rs2[w]; }
        rs[0] = a; rs2[0] = a2;
    }
    __syncthreads();
    const float mean = rs[0] * (1.0f / Dm);
    const float var  = rs2[0] * (1.0f / Dm) - mean * mean;
    const float inv  = rsqrtf(var + 1e-5f);
    for (int i = threadIdx.x; i < Dm; i += 256)
        y[i] = (y[i] - mean) * inv * gam[i] + bet[i];
}

// ---------------- host launcher ----------------
extern "C" void kernel_launch(void* const* d_in, const int* in_sizes, int n_in,
                              void* d_out, int out_size, void* d_ws, size_t ws_size,
                              hipStream_t stream) {
    (void)in_sizes; (void)n_in; (void)out_size; (void)ws_size;
    const float* x     = (const float*)d_in[0];
    const float* Ww    = (const float*)d_in[1];
    const float* Wb    = (const float*)d_in[2];
    const float* projw = (const float*)d_in[3];
    const float* projb = (const float*)d_in[4];
    const float* lng   = (const float*)d_in[5];
    const float* lnb   = (const float*)d_in[6];
    const float* rpb   = (const float*)d_in[7];
    float* out = (float*)d_out;

    // workspace layout (bf16 elements)
    const size_t N_X  = (size_t)Mrows * Dm;         // 4,194,304
    const size_t N_W  = (size_t)3 * Dm * Dm;        // 3,145,728
    const size_t N_P  = (size_t)Dm * Dm;            // 1,048,576
    __bf16* ws   = (__bf16*)d_ws;
    __bf16* xbf  = ws;
    __bf16* Wbf  = xbf + N_X;
    __bf16* Pbf  = Wbf + N_W;
    __bf16* qbf  = Pbf + N_P;
    __bf16* kbf  = qbf + N_X;
    __bf16* vTbf = kbf + N_X;
    __bf16* obf  = vTbf + N_X;

    cvt_f32_bf16<<<(int)((N_X + 255) / 256), 256, 0, stream>>>(x, xbf, (int)N_X);
    cvt_f32_bf16<<<(int)((N_W + 255) / 256), 256, 0, stream>>>(Ww, Wbf, (int)N_W);
    cvt_f32_bf16<<<(int)((N_P + 255) / 256), 256, 0, stream>>>(projw, Pbf, (int)N_P);

    // QKV: 128 m-tiles(32 rows) * 48 n-tiles = 6144 waves -> 768 blocks of 8 waves
    qkv_gemm<<<768, 256, 0, stream>>>(xbf, Wbf, Wb, qbf, kbf, vTbf);

    // attention: B*H*(S/16) = 4096 single-wave blocks
    attn_kernel<<<4096, 32, 0, stream>>>(qbf, kbf, vTbf, rpb, obf);

    // projection: 128 m-tiles(32 rows) * 16 n-tiles = 2048 waves -> 256 blocks
    proj_gemm<<<256, 256, 0, stream>>>(obf, Pbf, projb, x, out);

    // layernorm: one block per row
    ln_kernel<<<Mrows, 256, 0, stream>>>(out, lng, lnb);
}
